// BNN_36532991820264
// MI455X (gfx1250) — matrix operand order repaired and checked by reading, not compiled, not run on hardware
//
#include <hip/hip_runtime.h>

// ---------------------------------------------------------------------------
// BNN predictive forward on gfx1250 (MI455X), bf16 WMMA path.
//   S=32 samples, B=1024, D_IN=784, D_H=1024, D_OUT=10
//   layer:  act = relu(act @ (we*exp(.5wv)+wm) + (be*exp(.5bv)+bm))
//   head :  logits = act @ Wl + bl
// ---------------------------------------------------------------------------

typedef __attribute__((ext_vector_type(4)))  __bf16 v4bf;
typedef __attribute__((ext_vector_type(8)))  __bf16 v8bf;
typedef __attribute__((ext_vector_type(16))) __bf16 v16bf;
typedef __attribute__((ext_vector_type(8)))  float  v8f;

#define NS    32
#define BATCH 1024
#define DIN   784
#define DH    1024
#define DOUT  10

__device__ __forceinline__ v16bf cat8(v8bf lo, v8bf hi) {
    return __builtin_shufflevector(lo, hi, 0,1,2,3,4,5,6,7,8,9,10,11,12,13,14,15);
}

// ---------------------------------------------------------------------------
// Hidden layer GEMM: out[s,b,n] = relu( sum_k A[s,b,k] * W[s,k,n] + bias[s,n] )
// A is f32 x (broadcast over s) when FIRST, else bf16 activations (S,B,K).
// Block tile: 128(M) x 128(N), K-step 32. 256 threads = 8 waves (4M x 2N).
// All staging loads are unconditional (clamped addresses) + select-zero:
// no divergent branches in the K loop.
// ---------------------------------------------------------------------------
template <bool FIRST, int K>
__global__ __launch_bounds__(256) void bnn_hidden_gemm(
    const float* __restrict__ xf,               // FIRST: (B,K) f32
    const __bf16* __restrict__ xb,              // !FIRST: (S,B,K) bf16
    const float* __restrict__ we,               // (S,K,DH)
    const float* __restrict__ wv,               // (K,DH)
    const float* __restrict__ wm,               // (K,DH)
    const float* __restrict__ bm,               // (DH)
    const float* __restrict__ bv,               // (DH)
    const float* __restrict__ be,               // (S,DH)
    __bf16* __restrict__ out)                   // (S,B,DH) bf16
{
    constexpr int N  = DH;
    constexpr int LS = 40;                      // LDS K-stride (halves), padded

    const int s    = blockIdx.z;
    const int m0   = blockIdx.y * 128;
    const int n0   = blockIdx.x * 128;
    const int tid  = threadIdx.x;
    const int lane = tid & 31;
    const int wv_id = tid >> 5;
    const int wavem = wv_id & 3;                // 0..3 -> 32-row slab
    const int waven = wv_id >> 2;               // 0..1 -> 64-col slab

    __shared__ __align__(16) __bf16 As[128 * LS];   // [m][k]
    __shared__ __align__(16) __bf16 Ws[128 * LS];   // [n][k] (transposed)

    v8f acc[2][4];
    #pragma unroll
    for (int i = 0; i < 2; ++i)
        #pragma unroll
        for (int j = 0; j < 4; ++j) acc[i][j] = {};

    const int rrow = lane & 15;
    const int half = lane >> 4;

    const size_t we_base = (size_t)s * K * N;

    for (int k0 = 0; k0 < K; k0 += 32) {
        // ================= issue ALL global loads first (no branches) ======
        v8bf   ra8[2];          // !FIRST path: 2 x 16B
        float4 raf[4];          // FIRST path:  4 x 16B
        bool   av[4];
        if constexpr (FIRST) {
            #pragma unroll
            for (int i = 0; i < 4; ++i) {
                int vec = tid + i * 256;            // 0..1023
                int ml  = vec >> 3;                 // row
                int k4  = (vec & 7) * 4;            // 0,4,..28
                int kg  = k0 + k4;
                av[i]   = (kg < K);
                int kgc = av[i] ? kg : (K - 4);     // clamped, 16B-aligned
                raf[i] = *reinterpret_cast<const float4*>(
                    xf + (size_t)(m0 + ml) * K + kgc);
            }
        } else {
            #pragma unroll
            for (int i = 0; i < 2; ++i) {
                int vec = tid + i * 256;            // 0..511
                int ml  = vec >> 2;                 // row
                int k8  = (vec & 3) * 8;            // 0,8,16,24
                ra8[i] = *reinterpret_cast<const v8bf*>(
                    xb + ((size_t)s * BATCH + (m0 + ml)) * K + (k0 + k8));
            }
        }
        // ---- W tile (32 x 128): sample we*exp(.5wv)+wm, float4 reads ----
        float4 rwe[4], rwv[4], rwm[4];
        bool   wvld[4];
        #pragma unroll
        for (int i = 0; i < 4; ++i) {
            int task = tid + i * 256;               // 0..1023
            int nl4  = (task & 31) * 4;             // 0,4,..124
            int kl   = task >> 5;                   // 0..31
            int kg   = k0 + kl;
            wvld[i]  = (kg < K);
            int kgc  = wvld[i] ? kg : (K - 1);
            size_t wi = (size_t)kgc * N + (n0 + nl4);
            rwe[i] = *reinterpret_cast<const float4*>(we + we_base + wi);
            rwv[i] = *reinterpret_cast<const float4*>(wv + wi);
            rwm[i] = *reinterpret_cast<const float4*>(wm + wi);
        }

        // ================= stores to LDS =================
        if constexpr (FIRST) {
            #pragma unroll
            for (int i = 0; i < 4; ++i) {
                int vec = tid + i * 256;
                int ml  = vec >> 3;
                int k4  = (vec & 7) * 4;
                v4bf h = { (__bf16)raf[i].x, (__bf16)raf[i].y,
                           (__bf16)raf[i].z, (__bf16)raf[i].w };
                h = av[i] ? h : v4bf{};
                *reinterpret_cast<v4bf*>(&As[ml * LS + k4]) = h;
            }
        } else {
            #pragma unroll
            for (int i = 0; i < 2; ++i) {
                int vec = tid + i * 256;
                int ml  = vec >> 2;
                int k8  = (vec & 3) * 8;
                *reinterpret_cast<v8bf*>(&As[ml * LS + k8]) = ra8[i];
            }
        }
        #pragma unroll
        for (int i = 0; i < 4; ++i) {
            int task = tid + i * 256;
            int nl4  = (task & 31) * 4;
            int kl   = task >> 5;
            float sc = wvld[i] ? 1.0f : 0.0f;       // select-zero, no branch
            float w0 = sc * fmaf(rwe[i].x, __expf(0.5f * rwv[i].x), rwm[i].x);
            float w1 = sc * fmaf(rwe[i].y, __expf(0.5f * rwv[i].y), rwm[i].y);
            float w2 = sc * fmaf(rwe[i].z, __expf(0.5f * rwv[i].z), rwm[i].z);
            float w3 = sc * fmaf(rwe[i].w, __expf(0.5f * rwv[i].w), rwm[i].w);
            Ws[(nl4 + 0) * LS + kl] = (__bf16)w0;
            Ws[(nl4 + 1) * LS + kl] = (__bf16)w1;
            Ws[(nl4 + 2) * LS + kl] = (__bf16)w2;
            Ws[(nl4 + 3) * LS + kl] = (__bf16)w3;
        }
        __syncthreads();

        // ================= fragments + WMMA =================
        v16bf a[2];
        #pragma unroll
        for (int i = 0; i < 2; ++i) {
            const __bf16* p = &As[(wavem * 32 + i * 16 + rrow) * LS + half * 8];
            v8bf lo = *reinterpret_cast<const v8bf*>(p);
            v8bf hi = *reinterpret_cast<const v8bf*>(p + 16);
            a[i] = cat8(lo, hi);
        }
        v16bf b[4];
        #pragma unroll
        for (int j = 0; j < 4; ++j) {
            const __bf16* p = &Ws[(waven * 64 + j * 16 + rrow) * LS + half * 8];
            v8bf lo = *reinterpret_cast<const v8bf*>(p);
            v8bf hi = *reinterpret_cast<const v8bf*>(p + 16);
            b[j] = cat8(lo, hi);
        }
        #pragma unroll
        for (int i = 0; i < 2; ++i)
            #pragma unroll
            for (int j = 0; j < 4; ++j)
                acc[i][j] = __builtin_amdgcn_wmma_f32_16x16x32_bf16(
                    false, a[i], false, b[j], (short)0, acc[i][j], false, false);
        __syncthreads();
    }

    // ---- epilogue: bias sample + ReLU + bf16 store ----
    #pragma unroll
    for (int j = 0; j < 4; ++j) {
        int col = n0 + waven * 64 + j * 16 + rrow;
        float bias = bm[col] + be[(size_t)s * N + col] * __expf(0.5f * bv[col]);
        #pragma unroll
        for (int i = 0; i < 2; ++i) {
            int rb = m0 + wavem * 32 + i * 16 + half * 8;
            #pragma unroll
            for (int r = 0; r < 8; ++r) {
                float v = acc[i][j][r] + bias;
                v = v > 0.0f ? v : 0.0f;
                out[((size_t)s * BATCH + (rb + r)) * N + col] = (__bf16)v;
            }
        }
    }
}

// ---------------------------------------------------------------------------
// Head: logits[s,b,o] = sum_k act[s,b,k]*Wl[s,k,o] + bl[s,o], o < 10 (pad->16).
// 256 threads = 8 waves, each wave one 16x16 tile; 128 rows per block.
// ---------------------------------------------------------------------------
__global__ __launch_bounds__(256) void bnn_head_gemm(
    const __bf16* __restrict__ xb,              // (S,B,DH) bf16
    const float* __restrict__ we,               // (S,DH,DOUT)
    const float* __restrict__ wvv,              // (DH,DOUT)
    const float* __restrict__ wmm,              // (DH,DOUT)
    const float* __restrict__ bm,               // (DOUT)
    const float* __restrict__ bv,               // (DOUT)
    const float* __restrict__ be,               // (S,DOUT)
    float* __restrict__ out)                    // (S,B,DOUT) f32
{
    constexpr int LS = 40;
    const int s    = blockIdx.y;
    const int m0   = blockIdx.x * 128;
    const int tid  = threadIdx.x;
    const int lane = tid & 31;
    const int wv_id = tid >> 5;

    __shared__ __align__(16) __bf16 As[128 * LS];
    __shared__ __align__(16) __bf16 Ws[16 * LS];

    v8f acc = {};
    const int rrow = lane & 15;
    const int half = lane >> 4;

    for (int k0 = 0; k0 < DH; k0 += 32) {
        // ---- issue global loads (branchless: clamp + select) ----
        v8bf ra8[2];
        #pragma unroll
        for (int i = 0; i < 2; ++i) {
            int vec = tid + i * 256;
            int ml  = vec >> 2;
            int k8  = (vec & 3) * 8;
            ra8[i] = *reinterpret_cast<const v8bf*>(
                xb + ((size_t)s * BATCH + (m0 + ml)) * DH + (k0 + k8));
        }
        float rwe[2], rwv[2], rwm[2];
        bool  nv[2];
        #pragma unroll
        for (int i = 0; i < 2; ++i) {
            int e  = tid + i * 256;                 // 0..511
            int kl = e >> 4, nl = e & 15;
            nv[i]  = (nl < DOUT);
            int nc = nv[i] ? nl : (DOUT - 1);       // clamped, in-bounds
            size_t wi = (size_t)(k0 + kl) * DOUT + nc;
            rwe[i] = we[(size_t)s * DH * DOUT + wi];
            rwv[i] = wvv[wi];
            rwm[i] = wmm[wi];
        }
        // ---- LDS stores ----
        #pragma unroll
        for (int i = 0; i < 2; ++i) {
            int vec = tid + i * 256;
            int ml  = vec >> 2;
            int k8  = (vec & 3) * 8;
            *reinterpret_cast<v8bf*>(&As[ml * LS + k8]) = ra8[i];
        }
        #pragma unroll
        for (int i = 0; i < 2; ++i) {
            int e  = tid + i * 256;
            int kl = e >> 4, nl = e & 15;
            float sc = nv[i] ? 1.0f : 0.0f;
            float w  = sc * fmaf(rwe[i], __expf(0.5f * rwv[i]), rwm[i]);
            Ws[nl * LS + kl] = (__bf16)w;
        }
        __syncthreads();

        const __bf16* pa = &As[(wv_id * 16 + rrow) * LS + half * 8];
        v16bf a = cat8(*reinterpret_cast<const v8bf*>(pa),
                       *reinterpret_cast<const v8bf*>(pa + 16));
        const __bf16* pb = &Ws[rrow * LS + half * 8];
        v16bf b = cat8(*reinterpret_cast<const v8bf*>(pb),
                       *reinterpret_cast<const v8bf*>(pb + 16));
        acc = __builtin_amdgcn_wmma_f32_16x16x32_bf16(
            false, a, false, b, (short)0, acc, false, false);
        __syncthreads();
    }

    if (rrow < DOUT) {
        float bias = bm[rrow] + be[(size_t)s * DOUT + rrow] * __expf(0.5f * bv[rrow]);
        int rb = m0 + wv_id * 16 + half * 8;
        #pragma unroll
        for (int r = 0; r < 8; ++r)
            out[((size_t)s * BATCH + (rb + r)) * DOUT + rrow] = acc[r] + bias;
    }
}

// ---------------------------------------------------------------------------
// Launch: 3 dependent GEMMs on `stream`. Workspace holds two bf16 activation
// tensors of (S,B,DH) = 64 MiB each.
// ---------------------------------------------------------------------------
extern "C" void kernel_launch(void* const* d_in, const int* in_sizes, int n_in,
                              void* d_out, int out_size, void* d_ws, size_t ws_size,
                              hipStream_t stream) {
    const float* x   = (const float*)d_in[0];
    const float* wm0 = (const float*)d_in[1];
    const float* wv0 = (const float*)d_in[2];
    const float* bm0 = (const float*)d_in[3];
    const float* bv0 = (const float*)d_in[4];
    const float* wm1 = (const float*)d_in[5];
    const float* wv1 = (const float*)d_in[6];
    const float* bm1 = (const float*)d_in[7];
    const float* bv1 = (const float*)d_in[8];
    const float* wlm = (const float*)d_in[9];
    const float* wlv = (const float*)d_in[10];
    const float* blm = (const float*)d_in[11];
    const float* blv = (const float*)d_in[12];
    const float* we0 = (const float*)d_in[13];
    const float* be0 = (const float*)d_in[14];
    const float* we1 = (const float*)d_in[15];
    const float* be1 = (const float*)d_in[16];
    const float* wel = (const float*)d_in[17];
    const float* bel = (const float*)d_in[18];

    __bf16* act1 = (__bf16*)d_ws;                          // (S,B,DH) bf16
    __bf16* act2 = act1 + (size_t)NS * BATCH * DH;         // (S,B,DH) bf16

    dim3 blk(256);
    dim3 grid_h(DH / 128, BATCH / 128, NS);   // (8, 8, 32)

    bnn_hidden_gemm<true, DIN><<<grid_h, blk, 0, stream>>>(
        x, nullptr, we0, wv0, wm0, bm0, bv0, be0, act1);

    bnn_hidden_gemm<false, DH><<<grid_h, blk, 0, stream>>>(
        nullptr, act1, we1, wv1, wm1, bm1, bv1, be1, act2);

    bnn_head_gemm<<<dim3(BATCH / 128, NS), blk, 0, stream>>>(
        act2, wel, wlv, wlm, blm, blv, bel, (float*)d_out);
}
